// MixtureOfExperts_56745107915274
// MI455X (gfx1250) — compile-verified, matrix operand order checked by reading, not run on hardware
//
#include <hip/hip_runtime.h>
#include <math.h>

// ---------------------------------------------------------------------------
// Problem constants (from the reference)
// ---------------------------------------------------------------------------
#define B_   8
#define S_   512
#define D_   512
#define H_   8
#define E_   64
#define DFF_ 2048
#define ACT_ELEMS (B_ * S_ * D_)            // 2,097,152
#define SCALE_ (0.125f)                     // 1/sqrt(64)

// ---------------------------------------------------------------------------
// WMMA types / helpers (CDNA5 gfx1250, wave32)
// ---------------------------------------------------------------------------
typedef __attribute__((ext_vector_type(16))) __bf16 v16bf;
typedef __attribute__((ext_vector_type(8)))  float  v8f;

union FragBF {
    v16bf v;
    unsigned short u[16];
    uint4 q[2];
};

__device__ __forceinline__ unsigned short f2bf(float f) {
    unsigned int u = __float_as_uint(f);
    unsigned int r = u + 0x7FFFu + ((u >> 16) & 1u);   // round-to-nearest-even
    return (unsigned short)(r >> 16);
}

__device__ __forceinline__ float gelu_exact(float x) {
    return 0.5f * x * (1.0f + erff(x * 0.70710678118654752440f));
}

__device__ __forceinline__ v8f wmma_bf16(const FragBF& a, const FragBF& b, v8f c) {
    return __builtin_amdgcn_wmma_f32_16x16x32_bf16(
        false, a.v, false, b.v, (short)0, c, false, false);
}

// ---------------------------------------------------------------------------
// f32 -> bf16 conversion (vectorized, 4 elems/thread). n must be %4 == 0.
// ---------------------------------------------------------------------------
__global__ void convert_bf16_k(const float* __restrict__ src,
                               unsigned short* __restrict__ dst, int n4) {
    int i = blockIdx.x * blockDim.x + threadIdx.x;
    if (i >= n4) return;
    float4 f = ((const float4*)src)[i];
    ushort4 u;
    u.x = f2bf(f.x); u.y = f2bf(f.y); u.z = f2bf(f.z); u.w = f2bf(f.w);
    ((ushort4*)dst)[i] = u;
}

// ---------------------------------------------------------------------------
// GEMM:  C[M,N] = act( A[M,K] @ W[N,K]^T + bias[N] )
// A, W pre-converted bf16; f32 accumulate/output.
// Tiles staged into LDS with CDNA5 async global->LDS copies (ASYNCcnt).
// Requires M%128==0, N%64==0, K%32==0 (true for all call sites).
// Block: 256 threads = 8 waves; block tile 128x64; wave tile 32x32.
// ---------------------------------------------------------------------------
#define GM_BM 128
#define GM_BN 64
#define GM_BK 32
#define LDS_STRIDE 48   // padded row stride (ushorts) -> 96B, 16B aligned

__launch_bounds__(256)
__global__ void gemm_bf16_wmma_k(const unsigned short* __restrict__ A,
                                 const unsigned short* __restrict__ W,
                                 const float* __restrict__ bias,
                                 float* __restrict__ C,
                                 int M, int N, int K, int act) {
    __shared__ __align__(16) unsigned short lA[GM_BM * LDS_STRIDE];
    __shared__ __align__(16) unsigned short lB[GM_BN * LDS_STRIDE];

    const int tid  = threadIdx.x;
    const int lane = tid & 31;
    const int wave = tid >> 5;
    const int half = lane >> 4;
    const int lr   = lane & 15;

    const int bM = blockIdx.y * GM_BM;
    const int bN = blockIdx.x * GM_BN;
    const int wm = (wave & 3) * 32;   // wave M offset in block
    const int wn = (wave >> 2) * 32;  // wave N offset in block

    // Async-copy addressing: 16B per lane per instruction.
    // B tile: 64 rows x 64B  -> 256 chunks -> 1 instruction (all 8 waves).
    // A tile: 128 rows x 64B -> 512 chunks -> 2 instructions.
    const int rowB  = tid >> 2;
    const int partB = tid & 3;
    const int rowA0 = tid >> 2;           // rows 0..63
    const int rowA1 = rowA0 + 64;         // rows 64..127
    const int partA = tid & 3;

    unsigned offB  = (unsigned)(((bN + rowB ) * K) * 2 + partB * 16);
    unsigned offA0 = (unsigned)(((bM + rowA0) * K) * 2 + partA * 16);
    unsigned offA1 = (unsigned)(((bM + rowA1) * K) * 2 + partA * 16);
    unsigned ldsB  = (unsigned)(unsigned long long)(&lB[rowB  * LDS_STRIDE + partB * 8]);
    unsigned ldsA0 = (unsigned)(unsigned long long)(&lA[rowA0 * LDS_STRIDE + partA * 8]);
    unsigned ldsA1 = (unsigned)(unsigned long long)(&lA[rowA1 * LDS_STRIDE + partA * 8]);

    v8f acc[2][2];
    for (int i = 0; i < 2; i++)
        for (int j = 0; j < 2; j++)
            for (int r = 0; r < 8; r++) acc[i][j][r] = 0.0f;

    const int kc0 = half * 8;
    const int kc1 = 16 + half * 8;

    for (int k0 = 0; k0 < K; k0 += GM_BK) {
        __syncthreads();   // previous iteration finished reading LDS
        asm volatile("global_load_async_to_lds_b128 %0, %1, %2"
                     :: "v"(ldsA0), "v"(offA0), "s"(A) : "memory");
        asm volatile("global_load_async_to_lds_b128 %0, %1, %2"
                     :: "v"(ldsA1), "v"(offA1), "s"(A) : "memory");
        asm volatile("global_load_async_to_lds_b128 %0, %1, %2"
                     :: "v"(ldsB),  "v"(offB),  "s"(W) : "memory");
        offA0 += 2 * GM_BK;   // advance 32 bf16 = 64 bytes
        offA1 += 2 * GM_BK;
        offB  += 2 * GM_BK;
        asm volatile("s_wait_asynccnt 0" ::: "memory");
        __syncthreads();   // all waves' async copies have landed in LDS

        FragBF a[2], b[2];
        for (int i = 0; i < 2; i++) {
            const unsigned short* p = &lA[(wm + i * 16 + lr) * LDS_STRIDE];
            a[i].q[0] = *(const uint4*)&p[kc0];
            a[i].q[1] = *(const uint4*)&p[kc1];
        }
        for (int j = 0; j < 2; j++) {
            const unsigned short* p = &lB[(wn + j * 16 + lr) * LDS_STRIDE];
            b[j].q[0] = *(const uint4*)&p[kc0];
            b[j].q[1] = *(const uint4*)&p[kc1];
        }
        for (int i = 0; i < 2; i++)
            for (int j = 0; j < 2; j++)
                acc[i][j] = wmma_bf16(a[i], b[j], acc[i][j]);
    }

    // Epilogue: C layout -> row = r + 8*half, col = lr
    for (int i = 0; i < 2; i++)
        for (int j = 0; j < 2; j++) {
            int col = bN + wn + j * 16 + lr;
            float bv = bias ? bias[col] : 0.0f;
            for (int r = 0; r < 8; r++) {
                int row = bM + wm + i * 16 + r + half * 8;
                float v = acc[i][j][r] + bv;
                if (act == 1) v = gelu_exact(v);
                C[(size_t)row * N + col] = v;
            }
        }
}

// ---------------------------------------------------------------------------
// Flash attention: per (b, head, 64-query tile). Block = 128 thr = 4 waves;
// each wave owns 16 query rows. Key chunks of 64, online softmax, QK^T and
// P*V via WMMA. Scale folded into Q at load time.
// ---------------------------------------------------------------------------
#define FA_PAD 72   // padded key/E row stride in ushorts (144B, 16B aligned)

__launch_bounds__(128)
__global__ void flash_attn_k(const float* __restrict__ q,
                             const float* __restrict__ kmat,
                             const float* __restrict__ vmat,
                             float* __restrict__ outp) {
    const int b  = blockIdx.z;
    const int hh = blockIdx.y;
    const int q0 = blockIdx.x * 64;

    const int tid  = threadIdx.x;
    const int lane = tid & 31;
    const int w    = tid >> 5;
    const int half = lane >> 4;
    const int lr   = lane & 15;
    const int myQ  = q0 + w * 16;

    __shared__ __align__(16) unsigned short lK[64 * FA_PAD];        // [key][E]
    __shared__ __align__(16) unsigned short lV[64 * FA_PAD];        // transposed: [E][key]
    __shared__ __align__(16) unsigned short lP[4][16 * FA_PAD];     // per-wave P staging

    const int kc0 = half * 8;
    const int kc1 = 16 + half * 8;

    // Q fragments (held in registers, scaled by 1/sqrt(E))
    FragBF qf[2];
    {
        const float* qp = q + ((size_t)(b * S_ + myQ + lr) * D_ + hh * E_);
        for (int ks = 0; ks < 2; ks++) {
            int c0 = ks * 32 + kc0;
            int c1 = ks * 32 + kc1;
            float4 f0 = *(const float4*)&qp[c0];
            float4 f1 = *(const float4*)&qp[c0 + 4];
            float4 f2 = *(const float4*)&qp[c1];
            float4 f3 = *(const float4*)&qp[c1 + 4];
            qf[ks].u[0]  = f2bf(f0.x * SCALE_); qf[ks].u[1]  = f2bf(f0.y * SCALE_);
            qf[ks].u[2]  = f2bf(f0.z * SCALE_); qf[ks].u[3]  = f2bf(f0.w * SCALE_);
            qf[ks].u[4]  = f2bf(f1.x * SCALE_); qf[ks].u[5]  = f2bf(f1.y * SCALE_);
            qf[ks].u[6]  = f2bf(f1.z * SCALE_); qf[ks].u[7]  = f2bf(f1.w * SCALE_);
            qf[ks].u[8]  = f2bf(f2.x * SCALE_); qf[ks].u[9]  = f2bf(f2.y * SCALE_);
            qf[ks].u[10] = f2bf(f2.z * SCALE_); qf[ks].u[11] = f2bf(f2.w * SCALE_);
            qf[ks].u[12] = f2bf(f3.x * SCALE_); qf[ks].u[13] = f2bf(f3.y * SCALE_);
            qf[ks].u[14] = f2bf(f3.z * SCALE_); qf[ks].u[15] = f2bf(f3.w * SCALE_);
        }
    }

    float mrow[8], lrow[8];
    v8f ov[4];
    for (int r = 0; r < 8; r++) { mrow[r] = -1e30f; lrow[r] = 0.0f; }
    for (int n = 0; n < 4; n++)
        for (int r = 0; r < 8; r++) ov[n][r] = 0.0f;

    for (int kc = 0; kc < S_ / 64; kc++) {
        __syncthreads();
        // Stage K chunk [key][E] and V chunk transposed [E][key]
        for (int c = tid; c < 64 * 16; c += 128) {
            int row = c >> 4;
            int col = (c & 15) << 2;
            size_t g = (size_t)(b * S_ + kc * 64 + row) * D_ + hh * E_ + col;
            float4 fk = *(const float4*)&kmat[g];
            unsigned short* dk = &lK[row * FA_PAD + col];
            dk[0] = f2bf(fk.x); dk[1] = f2bf(fk.y); dk[2] = f2bf(fk.z); dk[3] = f2bf(fk.w);
            float4 fv = *(const float4*)&vmat[g];
            lV[(col + 0) * FA_PAD + row] = f2bf(fv.x);
            lV[(col + 1) * FA_PAD + row] = f2bf(fv.y);
            lV[(col + 2) * FA_PAD + row] = f2bf(fv.z);
            lV[(col + 3) * FA_PAD + row] = f2bf(fv.w);
        }
        __syncthreads();

        // S = (scaled Q) . K^T   (4 key tiles x 2 K-steps)
        v8f s[4];
        for (int n = 0; n < 4; n++)
            for (int r = 0; r < 8; r++) s[n][r] = 0.0f;
        for (int ks = 0; ks < 2; ks++) {
            for (int n = 0; n < 4; n++) {
                FragBF bk;
                const unsigned short* p = &lK[(n * 16 + lr) * FA_PAD + ks * 32];
                bk.q[0] = *(const uint4*)&p[kc0];
                bk.q[1] = *(const uint4*)&p[kc1];
                s[n] = wmma_bf16(qf[ks], bk, s[n]);
            }
        }

        // Online softmax (row reductions across 16-lane halves)
        float corr[8];
        for (int r = 0; r < 8; r++) {
            float t = fmaxf(fmaxf(s[0][r], s[1][r]), fmaxf(s[2][r], s[3][r]));
            t = fmaxf(t, __shfl_xor(t, 1));
            t = fmaxf(t, __shfl_xor(t, 2));
            t = fmaxf(t, __shfl_xor(t, 4));
            t = fmaxf(t, __shfl_xor(t, 8));
            float mn = fmaxf(mrow[r], t);
            corr[r] = __expf(mrow[r] - mn);
            mrow[r] = mn;
            float rs = 0.0f;
            for (int n = 0; n < 4; n++) {
                float e = __expf(s[n][r] - mn);
                s[n][r] = e;
                rs += e;
            }
            rs += __shfl_xor(rs, 1);
            rs += __shfl_xor(rs, 2);
            rs += __shfl_xor(rs, 4);
            rs += __shfl_xor(rs, 8);
            lrow[r] = lrow[r] * corr[r] + rs;
        }
        for (int n = 0; n < 4; n++)
            for (int r = 0; r < 8; r++) ov[n][r] *= corr[r];

        // P (C layout) -> per-wave LDS -> A-fragment layout
        unsigned short* pw = lP[w];
        for (int n = 0; n < 4; n++)
            for (int r = 0; r < 8; r++)
                pw[(r + 8 * half) * FA_PAD + n * 16 + lr] = f2bf(s[n][r]);
        asm volatile("s_wait_dscnt 0" ::: "memory");   // wave-local LDS RAW

        // O += P . V   (A = P [16 x 64 keys], B = V^T tiles)
        for (int ks = 0; ks < 2; ks++) {
            FragBF ap;
            const unsigned short* pp = &pw[lr * FA_PAD + ks * 32];
            ap.q[0] = *(const uint4*)&pp[kc0];
            ap.q[1] = *(const uint4*)&pp[kc1];
            for (int n = 0; n < 4; n++) {
                FragBF bv;
                const unsigned short* pv = &lV[(n * 16 + lr) * FA_PAD + ks * 32];
                bv.q[0] = *(const uint4*)&pv[kc0];
                bv.q[1] = *(const uint4*)&pv[kc1];
                ov[n] = wmma_bf16(ap, bv, ov[n]);
            }
        }
    }

    // Normalize and store
    for (int n = 0; n < 4; n++) {
        int col = hh * E_ + n * 16 + lr;
        for (int r = 0; r < 8; r++) {
            int row = myQ + r + 8 * half;
            outp[(size_t)(b * S_ + row) * D_ + col] = ov[n][r] / lrow[r];
        }
    }
}

// ---------------------------------------------------------------------------
// Row LayerNorm: out = act( LN(x + res) * g + b ), one block per row.
// ---------------------------------------------------------------------------
__launch_bounds__(128)
__global__ void layernorm_k(const float* __restrict__ x,
                            const float* __restrict__ res,
                            const float* __restrict__ g,
                            const float* __restrict__ bta,
                            float* __restrict__ out, int W, int act) {
    const int row = blockIdx.x;
    const float* xr = x + (size_t)row * W;
    const float* rr = res ? res + (size_t)row * W : nullptr;
    __shared__ float red[4];

    float s = 0.0f;
    for (int c = threadIdx.x; c < W; c += blockDim.x)
        s += xr[c] + (rr ? rr[c] : 0.0f);
    for (int m = 16; m >= 1; m >>= 1) s += __shfl_xor(s, m);
    if ((threadIdx.x & 31) == 0) red[threadIdx.x >> 5] = s;
    __syncthreads();
    float mean = (red[0] + red[1] + red[2] + red[3]) / (float)W;

    float vs = 0.0f;
    for (int c = threadIdx.x; c < W; c += blockDim.x) {
        float v = xr[c] + (rr ? rr[c] : 0.0f) - mean;
        vs += v * v;
    }
    for (int m = 16; m >= 1; m >>= 1) vs += __shfl_xor(vs, m);
    __syncthreads();
    if ((threadIdx.x & 31) == 0) red[threadIdx.x >> 5] = vs;
    __syncthreads();
    float inv = rsqrtf((red[0] + red[1] + red[2] + red[3]) / (float)W + 1e-5f);

    for (int c = threadIdx.x; c < W; c += blockDim.x) {
        float v = xr[c] + (rr ? rr[c] : 0.0f);
        float y = (v - mean) * inv * g[c] + bta[c];
        if (act == 1) y = gelu_exact(y);
        out[(size_t)row * W + c] = y;
    }
}

// ---------------------------------------------------------------------------
// Small linear (gating MLPs, 8 rows): out[r,n] = act(in[r]·W[n] + b[n])
// act: 0 none, 1 tanh, 2 sigmoid, 3 relu, 4 gelu
// ---------------------------------------------------------------------------
__global__ void small_linear_k(const float* __restrict__ in,
                               const float* __restrict__ w,
                               const float* __restrict__ b,
                               float* __restrict__ out,
                               int R, int N, int K, int act) {
    int idx = blockIdx.x * blockDim.x + threadIdx.x;
    if (idx >= R * N) return;
    int r = idx / N, n = idx % N;
    const float* ir = in + (size_t)r * K;
    const float* wr = w + (size_t)n * K;
    float s = b ? b[n] : 0.0f;
    for (int kk = 0; kk < K; kk++) s += ir[kk] * wr[kk];
    if (act == 1) s = tanhf(s);
    else if (act == 2) s = 1.0f / (1.0f + __expf(-s));
    else if (act == 3) s = fmaxf(s, 0.0f);
    else if (act == 4) s = gelu_exact(s);
    out[idx] = s;
}

__global__ void elemmul_k(const float* __restrict__ a, const float* __restrict__ b,
                          float* __restrict__ c, int n) {
    int i = blockIdx.x * blockDim.x + threadIdx.x;
    if (i < n) c[i] = a[i] * b[i];
}

// ---------------------------------------------------------------------------
// Router: temperature, group softmax, combine, softmax, top-2, renorm,
// scatter, *0.7; writes sw [8,3], full [8,4] and load_loss.
// ---------------------------------------------------------------------------
__global__ void router_k(const float* __restrict__ logits_raw,
                         const float* __restrict__ grp_logits,
                         const float* __restrict__ temp,
                         float* __restrict__ sw,
                         float* __restrict__ out_loss,
                         float* __restrict__ out_full) {
    if (threadIdx.x != 0 || blockIdx.x != 0) return;
    float t = fminf(fmaxf(temp[0], 0.1f), 5.0f);
    float usage[3] = {0.f, 0.f, 0.f};
    for (int bi = 0; bi < B_; bi++) {
        float lg[3];
        for (int i = 0; i < 3; i++) lg[i] = logits_raw[bi * 3 + i] / t;
        float g0 = grp_logits[bi * 2 + 0], g1 = grp_logits[bi * 2 + 1];
        float gm = fmaxf(g0, g1);
        float e0 = __expf(g0 - gm), e1 = __expf(g1 - gm);
        float gs = e0 + e1;
        float ge[3] = {e0 / gs, e1 / gs, e0 / gs};
        float comb[3];
        for (int i = 0; i < 3; i++) comb[i] = lg[i] + 0.5f * __logf(ge[i] + 1e-8f);
        float cm = fmaxf(comb[0], fmaxf(comb[1], comb[2]));
        float ex[3], ssum = 0.f;
        for (int i = 0; i < 3; i++) { ex[i] = __expf(comb[i] - cm); ssum += ex[i]; }
        float smx[3];
        for (int i = 0; i < 3; i++) smx[i] = ex[i] / ssum;
        int i0 = 0;
        for (int i = 1; i < 3; i++) if (smx[i] > smx[i0]) i0 = i;
        int i1 = -1;
        for (int i = 0; i < 3; i++) {
            if (i == i0) continue;
            if (i1 < 0 || smx[i] > smx[i1]) i1 = i;
        }
        float a = smx[i0], bb = smx[i1];
        float mm = fmaxf(a, bb);
        float ea = __expf(a - mm), eb = __expf(bb - mm);
        float es = ea + eb;
        float o[3] = {0.f, 0.f, 0.f};
        o[i0] = ea / es; o[i1] = eb / es;
        out_full[bi * 4 + 0] = 0.3f;
        for (int i = 0; i < 3; i++) {
            o[i] *= 0.7f;
            sw[bi * 3 + i] = o[i];
            usage[i] += o[i] / (float)B_;
            out_full[bi * 4 + 1 + i] = o[i];
        }
    }
    float ll = 0.f;
    for (int i = 0; i < 3; i++) {
        float d = usage[i] - (1.0f / 3.0f);
        ll += d * d;
    }
    out_loss[0] = ll / 3.0f;
}

// ---------------------------------------------------------------------------
// Mixture accumulate: e==0 -> mixed = 0.3*eo ; else mixed += sw[b,e-1]*
// (scale[b,d]*eo + shift[b,d])
// ---------------------------------------------------------------------------
__global__ void accum_k(const float* __restrict__ eo, const float* __restrict__ sw,
                        const float* __restrict__ scl, const float* __restrict__ shf,
                        float* __restrict__ mixed, int expert) {
    size_t idx = (size_t)blockIdx.x * blockDim.x + threadIdx.x;
    if (idx >= (size_t)ACT_ELEMS) return;
    int bi = (int)(idx / (S_ * D_));
    int d = (int)(idx % D_);
    float e = eo[idx];
    if (expert == 0) {
        mixed[idx] = 0.3f * e;
    } else {
        float wv = sw[bi * 3 + (expert - 1)];
        mixed[idx] += wv * (scl[bi * D_ + d] * e + shf[bi * D_ + d]);
    }
}

// ---------------------------------------------------------------------------
// Orchestration
// ---------------------------------------------------------------------------
extern "C" void kernel_launch(void* const* d_in, const int* in_sizes, int n_in,
                              void* d_out, int out_size, void* d_ws, size_t ws_size,
                              hipStream_t stream) {
    (void)in_sizes; (void)n_in; (void)out_size; (void)ws_size;
    const float* x   = (const float*)d_in[0];
    const float* cov = (const float*)d_in[1];
    auto P = [&](int i) -> const float* { return (const float*)d_in[i]; };

    // Param leaf index map (jax tree_flatten: dict keys sorted alphabetically)
    // 2..5:    cov_imp  [w0,b0,w1,b1]
    // 6+e*36:  expert e; layer l at +l*16 with leaf order:
    //   0 bk, 1 bo, 2 bq, 3 bv, 4 c1b, 5 c1w, 6 c2b, 7 c2w,
    //   8 ln1_b, 9 ln1_g, 10 ln2_b, 11 ln2_g, 12 wk, 13 wo, 14 wq, 15 wv
    //   +32 norm_b, +33 norm_g, +34 proj_b, +35 proj_w
    // 150: grp_l1(w,b) 152: grp_l2 154: scale_l1 156: scale_l2
    // 158: shift_l1 160: shift_l2 162: spec_l1 164: spec_l2 166: spec_l3
    // 168: spec_ln1(g,b) 170: spec_ln2(g,b) 172: temp

    float* ws = (float*)d_ws;
    const size_t A = (size_t)ACT_ELEMS;
    float* h    = ws;
    float* qb   = ws + 1 * A;
    float* kb   = ws + 2 * A;
    float* vb   = ws + 3 * A;
    float* att  = ws + 4 * A;
    float* aob  = ws + 5 * A;
    float* eob  = ws + 6 * A;
    float* yff  = ws + 7 * A;   // 4*A elems (B*S*DFF)
    float* gz   = ws + 11 * A;
    float* wc     = gz;
    float* tb     = gz + 4096;
    float* impb   = gz + 8192;
    float* gA     = gz + 12288;   // 8x512
    float* gB     = gz + 16384;   // 8x256
    float* logitb = gz + 18432;   // 8x3
    float* grpH   = gz + 18464;   // 8x128
    float* grpL   = gz + 19488;   // 8x2
    float* scaleF = gz + 19520;   // 8x512
    float* shiftF = gz + 23616;   // 8x512
    float* tmpS   = gz + 27712;   // 8x256 scratch
    float* swb    = gz + 31808;   // 8x3
    // bf16 staging buffers (16B-aligned offsets)
    unsigned short* wbf = (unsigned short*)(ws + 11 * A + 65536);            // <=1,048,576 elems
    unsigned short* abf = (unsigned short*)(ws + 11 * A + 65536 + 524288);   // <=8,388,608 elems

    float* mixed = (float*)d_out;
    float* loss  = mixed + A;
    float* full  = loss + 1;

    auto CONV = [&](const float* s, unsigned short* d, int n) {
        convert_bf16_k<<<dim3((n / 4 + 255) / 256), 256, 0, stream>>>(s, d, n / 4);
    };

    // ----- gating path (tiny GEMMs, VALU) -----
    small_linear_k<<<dim3((8 * 512 + 127) / 128), 128, 0, stream>>>(cov, P(2), P(3), tb, 8, 512, 512, 1);
    small_linear_k<<<dim3((8 * 512 + 127) / 128), 128, 0, stream>>>(tb, P(4), P(5), impb, 8, 512, 512, 2);
    elemmul_k<<<dim3((8 * 512 + 127) / 128), 128, 0, stream>>>(cov, impb, wc, 8 * 512);
    small_linear_k<<<dim3((8 * 512 + 127) / 128), 128, 0, stream>>>(wc, P(162), P(163), gA, 8, 512, 512, 0);
    layernorm_k<<<8, 128, 0, stream>>>(gA, nullptr, P(168), P(169), gA, 512, 1);
    small_linear_k<<<dim3((8 * 256 + 127) / 128), 128, 0, stream>>>(gA, P(164), P(165), gB, 8, 256, 512, 0);
    layernorm_k<<<8, 128, 0, stream>>>(gB, nullptr, P(170), P(171), gB, 256, 1);
    small_linear_k<<<dim3((8 * 3 + 127) / 128), 128, 0, stream>>>(gB, P(166), P(167), logitb, 8, 3, 256, 0);
    small_linear_k<<<dim3((8 * 128 + 127) / 128), 128, 0, stream>>>(wc, P(150), P(151), grpH, 8, 128, 512, 3);
    small_linear_k<<<dim3((8 * 2 + 127) / 128), 128, 0, stream>>>(grpH, P(152), P(153), grpL, 8, 2, 128, 0);
    small_linear_k<<<dim3((8 * 256 + 127) / 128), 128, 0, stream>>>(wc, P(154), P(155), tmpS, 8, 256, 512, 3);
    small_linear_k<<<dim3((8 * 512 + 127) / 128), 128, 0, stream>>>(tmpS, P(156), P(157), scaleF, 8, 512, 256, 2);
    small_linear_k<<<dim3((8 * 256 + 127) / 128), 128, 0, stream>>>(wc, P(158), P(159), tmpS, 8, 256, 512, 3);
    small_linear_k<<<dim3((8 * 512 + 127) / 128), 128, 0, stream>>>(tmpS, P(160), P(161), shiftF, 8, 512, 256, 0);
    router_k<<<1, 32, 0, stream>>>(logitb, grpL, P(172), swb, loss, full);

    // ----- expert forwards (WMMA path, async-LDS staged bf16 GEMMs) -----
    const int M = B_ * S_;   // 4096
    auto GEMM = [&](const float* Wf, const float* bias, float* Cout,
                    int N, int K, int act) {
        CONV(Wf, wbf, N * K);
        gemm_bf16_wmma_k<<<dim3(N / GM_BN, M / GM_BM), 256, 0, stream>>>(
            abf, wbf, bias, Cout, M, N, K, act);
    };

    for (int e = 0; e < 4; e++) {
        const int ebase = 6 + e * 36;
        hipMemcpyAsync(h, x, A * sizeof(float), hipMemcpyDeviceToDevice, stream);
        for (int l = 0; l < 2; l++) {
            const int lb = ebase + l * 16;
            CONV(h, abf, M * D_);                     // shared by q/k/v GEMMs
            GEMM(P(lb + 14), P(lb + 2), qb, D_, D_, 0);
            GEMM(P(lb + 12), P(lb + 0), kb, D_, D_, 0);
            GEMM(P(lb + 15), P(lb + 3), vb, D_, D_, 0);
            flash_attn_k<<<dim3(S_ / 64, H_, B_), 128, 0, stream>>>(qb, kb, vb, att);
            CONV(att, abf, M * D_);
            GEMM(P(lb + 13), P(lb + 1), aob, D_, D_, 0);
            layernorm_k<<<M, 128, 0, stream>>>(h, aob, P(lb + 9), P(lb + 8), h, D_, 0);
            CONV(h, abf, M * D_);
            GEMM(P(lb + 5), P(lb + 4), yff, DFF_, D_, 1);
            CONV(yff, abf, M * DFF_);
            GEMM(P(lb + 7), P(lb + 6), aob, D_, DFF_, 0);
            layernorm_k<<<M, 128, 0, stream>>>(h, aob, P(lb + 11), P(lb + 10), h, D_, 0);
        }
        layernorm_k<<<M, 128, 0, stream>>>(h, nullptr, P(ebase + 33), P(ebase + 32), h, D_, 0);
        CONV(h, abf, M * D_);
        GEMM(P(ebase + 35), P(ebase + 34), eob, D_, D_, 0);
        accum_k<<<dim3((ACT_ELEMS + 255) / 256), 256, 0, stream>>>(eob, swb, scaleF, shiftF, mixed, e);
    }
}
